// HybridSSMTransformer_12163347382743
// MI455X (gfx1250) — matrix operand order repaired and checked by reading, not compiled, not run on hardware
//
#include <hip/hip_runtime.h>
#include <math.h>
#include <stdint.h>

// ---------------------------------------------------------------------------
// Hybrid SSM + Transformer forward pass for MI455X (gfx1250, wave32, WMMA).
// f16 activations/weights, fp32 accumulation via v_wmma_f32_16x16x32_f16.
// B tiles double-buffered in LDS via GLOBAL_LOAD_ASYNC_TO_LDS_B128 (ASYNCcnt),
// overlapping the async copy of tile k+1 with WMMA compute on tile k.
// ---------------------------------------------------------------------------

typedef __attribute__((ext_vector_type(16))) _Float16 v16h;
typedef __attribute__((ext_vector_type(8)))  _Float16 v8h;
typedef __attribute__((ext_vector_type(8)))  float    v8f;
typedef __attribute__((ext_vector_type(4)))  int      i4;

#define DEV static __device__ __forceinline__

#if defined(__has_builtin)
# if __has_builtin(__builtin_amdgcn_global_load_async_to_lds_b128) && \
     __has_builtin(__builtin_amdgcn_s_wait_asynccnt)
#  define USE_ASYNC_LDS 1
# endif
#endif
#ifndef USE_ASYNC_LDS
# define USE_ASYNC_LDS 0
#endif

constexpr int T_TOK = 2048;   // B * L
constexpr int NB    = 2;
constexpr int SEQ   = 1024;
constexpr int DM    = 512;
constexpr int DI    = 1024;
constexpr int NH    = 8;
constexpr int FF    = 2048;

// ---- workspace layout (bytes) ----
constexpr size_t algn(size_t x) { return (x + 255) & ~(size_t)255; }
constexpr size_t B_X    = 0;                                   // f32 T*DM
constexpr size_t B_XZ   = algn(B_X   + (size_t)T_TOK*DM*4);    // f32 T*2DI
constexpr size_t B_UC   = algn(B_XZ  + (size_t)T_TOK*2*DI*4);  // f32 T*DI
constexpr size_t B_DL   = algn(B_UC  + (size_t)T_TOK*DI*4);    // f32 T*DI
constexpr size_t B_BC   = algn(B_DL  + (size_t)T_TOK*DI*4);    // f32 T*32
constexpr size_t B_LN16 = algn(B_BC  + (size_t)T_TOK*32*4);    // f16 T*DM
constexpr size_t B_UC16 = algn(B_LN16+ (size_t)T_TOK*DM*2);    // f16 T*DI
constexpr size_t B_YM16 = algn(B_UC16+ (size_t)T_TOK*DI*2);    // f16 T*DI
constexpr size_t B_X16  = algn(B_YM16+ (size_t)T_TOK*DI*2);    // f16 T*DM
constexpr size_t B_TMP16= algn(B_X16 + (size_t)T_TOK*DM*2);    // f16 T*FF
constexpr size_t B_ATT16= algn(B_TMP16+(size_t)T_TOK*FF*2);    // f16 T*DM
constexpr size_t B_W16  = algn(B_ATT16+(size_t)T_TOK*DM*2);    // f16 per-layer weights

// per-layer f16 weight sub-offsets (element counts)
constexpr size_t W_INPROJ  = 0;
constexpr size_t W_DT      = W_INPROJ  + (size_t)2*DI*DM;
constexpr size_t W_XPROJ   = W_DT      + (size_t)DI*DI;
constexpr size_t W_OUTPROJ = W_XPROJ   + (size_t)32*DI;
constexpr size_t W_ATTNIN  = W_OUTPROJ + (size_t)DM*DI;
constexpr size_t W_ATTNOUT = W_ATTNIN  + (size_t)3*DM*DM;
constexpr size_t W_FFN1    = W_ATTNOUT + (size_t)DM*DM;
constexpr size_t W_FFN2    = W_FFN1    + (size_t)FF*DM;
constexpr size_t W_TOTAL   = W_FFN2    + (size_t)DM*FF;
constexpr size_t WS_BYTES  = B_W16 + W_TOTAL*2;

DEV v8f wmma_f32_f16(v16h a, v16h b, v8f c) {
    return __builtin_amdgcn_wmma_f32_16x16x32_f16(false, a, false, b,
                                                  (short)0, c, false, false);
}
DEV v16h make16(v8h lo, v8h hv) {
    v16h r;
#pragma unroll
    for (int i = 0; i < 8; ++i) { r[i] = lo[i]; r[8 + i] = hv[i]; }
    return r;
}
DEV float sigmoidf_(float x) { return 1.0f / (1.0f + __expf(-x)); }

#if USE_ASYNC_LDS
DEV void async_copy_b128(const _Float16* gsrc, _Float16* lds_dst) {
    __builtin_amdgcn_global_load_async_to_lds_b128(
        (__attribute__((address_space(1))) i4*)(uintptr_t)gsrc,
        (__attribute__((address_space(3))) i4*)(unsigned)(uintptr_t)lds_dst,
        0, 0);
}
#endif

// ---------------------------------------------------------------------------
// f32 -> f16 conversion
// ---------------------------------------------------------------------------
__global__ void __launch_bounds__(256)
cvt16_k(const float* __restrict__ in, _Float16* __restrict__ out, int n)
{
    const int i = blockIdx.x * 256 + threadIdx.x;
    if (i < n) out[i] = (_Float16)in[i];
}

// ---------------------------------------------------------------------------
// GEMM: C[M,N] = act(A16[M,K] * W16[N,K]^T + bias[N]) + res[M,N]
// act: 0 none, 1 gelu, 2 softplus.  Block = 128 thr (4 waves), tile 128x64.
// Wave tile 32x64: 2 A-frags x 4 B-frags -> 8 WMMA per 32-k step.
// B tile (64x32 f16 = 4KB) double-buffered in LDS; async copy of tile k+1
// overlaps WMMA compute on tile k; one barrier per k-step.
// ---------------------------------------------------------------------------
__global__ void __launch_bounds__(128)
gemm_wmma_k(const _Float16* __restrict__ A16, const _Float16* __restrict__ W16,
            const float* __restrict__ bias, const float* __restrict__ res,
            float* __restrict__ C, _Float16* __restrict__ C16,
            int K, int N, int act)
{
    __shared__ _Float16 bs[2][64][32];

    const int tid  = threadIdx.x;
    const int lane = tid & 31;
    const int wave = tid >> 5;
    const int ml   = lane & 15;
    const int hi   = lane >> 4;
    const int m0   = blockIdx.y * 128 + wave * 32;
    const int n0   = blockIdx.x * 64;

    // staging map: thread -> (row sn, 16-half chunk sq)
    const int sn  = tid & 63;
    const int sq  = tid >> 6;
    const int nld = (n0 + sn < N) ? (n0 + sn) : 0;   // clamp, no divergence
    const _Float16* wsrc = W16 + (size_t)nld * K + sq * 16;
    _Float16* sdst[2] = { &bs[0][sn][sq * 16], &bs[1][sn][sq * 16] };

    v8f acc[4][2];
#pragma unroll
    for (int j = 0; j < 4; ++j) { acc[j][0] = v8f{}; acc[j][1] = v8f{}; }

    const _Float16* a0p = A16 + (size_t)(m0 + ml) * K + hi * 8;
    const _Float16* a1p = A16 + (size_t)(m0 + 16 + ml) * K + hi * 8;

    // prologue: stage tile 0 into buffer 0
#if USE_ASYNC_LDS
    async_copy_b128(wsrc, sdst[0]);
    async_copy_b128(wsrc + 8, sdst[0] + 8);
    __builtin_amdgcn_s_wait_asynccnt(0);
#else
    *(v8h*)(sdst[0])     = *(const v8h*)(wsrc);
    *(v8h*)(sdst[0] + 8) = *(const v8h*)(wsrc + 8);
#endif
    __syncthreads();

    int cur = 0;
    for (int k0 = 0; k0 < K; k0 += 32) {
        // issue async copy of tile k0+32 into the other buffer (overlaps WMMA)
        if (k0 + 32 < K) {
            const _Float16* wsn = wsrc + k0 + 32;
            _Float16* dn = sdst[cur ^ 1];
#if USE_ASYNC_LDS
            async_copy_b128(wsn, dn);
            async_copy_b128(wsn + 8, dn + 8);
#else
            *(v8h*)(dn)     = *(const v8h*)(wsn);
            *(v8h*)(dn + 8) = *(const v8h*)(wsn + 8);
#endif
        }

        const v16h a0 = make16(*(const v8h*)(a0p + k0), *(const v8h*)(a0p + k0 + 16));
        const v16h a1 = make16(*(const v8h*)(a1p + k0), *(const v8h*)(a1p + k0 + 16));
        if (k0 + 32 < K) {
            __builtin_prefetch(a0p + k0 + 32, 0, 3);
            __builtin_prefetch(a1p + k0 + 32, 0, 3);
        }

#pragma unroll
        for (int j = 0; j < 4; ++j) {
            const _Float16* bp = &bs[cur][j * 16 + ml][hi * 8];
            const v16h b = make16(*(const v8h*)bp, *(const v8h*)(bp + 16));
            acc[j][0] = wmma_f32_f16(a0, b, acc[j][0]);
            acc[j][1] = wmma_f32_f16(a1, b, acc[j][1]);
        }

        // next buffer ready + all waves done reading this one
#if USE_ASYNC_LDS
        __builtin_amdgcn_s_wait_asynccnt(0);
#endif
        __syncthreads();
        cur ^= 1;
    }

    // epilogue: lane holds column ml; VGPR v holds row hi*8+v (per 16-row subtile)
#pragma unroll
    for (int j = 0; j < 4; ++j) {
        const int n = n0 + j * 16 + ml;
        if (n >= N) continue;
        const float bv = bias ? bias[n] : 0.0f;
#pragma unroll
        for (int mi = 0; mi < 2; ++mi) {
#pragma unroll
            for (int v = 0; v < 8; ++v) {
                const int row = m0 + mi * 16 + hi * 8 + v;
                float x = acc[j][mi][v] + bv;
                if (act == 1)      x = 0.5f * x * (1.0f + erff(x * 0.70710678118f));
                else if (act == 2) x = (x > 20.0f) ? x : log1pf(__expf(x));
                if (res) x += res[(size_t)row * N + n];
                if (C)   C[(size_t)row * N + n]   = x;
                if (C16) C16[(size_t)row * N + n] = (_Float16)x;
            }
        }
    }
}

// ---------------------------------------------------------------------------
// LayerNorm: one wave per token; optional f32 and f16 outputs
// ---------------------------------------------------------------------------
__global__ void __launch_bounds__(32)
layernorm_k(const float* __restrict__ in, float* __restrict__ out,
            _Float16* __restrict__ out16,
            const float* __restrict__ w, const float* __restrict__ b)
{
    const int t    = blockIdx.x;
    const int lane = threadIdx.x;
    const float* x = in + (size_t)t * DM;

    float vals[16];
    float s = 0.0f;
#pragma unroll
    for (int i = 0; i < 16; ++i) { vals[i] = x[lane + i * 32]; s += vals[i]; }
#pragma unroll
    for (int m = 1; m < 32; m <<= 1) s += __shfl_xor(s, m, 32);
    const float mean = s * (1.0f / DM);

    float vs = 0.0f;
#pragma unroll
    for (int i = 0; i < 16; ++i) { const float d = vals[i] - mean; vs += d * d; }
#pragma unroll
    for (int m = 1; m < 32; m <<= 1) vs += __shfl_xor(vs, m, 32);
    const float inv = rsqrtf(vs * (1.0f / DM) + 1e-5f);

#pragma unroll
    for (int i = 0; i < 16; ++i) {
        const int c = lane + i * 32;
        const float y = (vals[i] - mean) * inv * w[c] + b[c];
        if (out)   out[(size_t)t * DM + c]   = y;
        if (out16) out16[(size_t)t * DM + c] = (_Float16)y;
    }
}

// ---------------------------------------------------------------------------
// Depthwise conv1d(k=3,pad=1) + bias + SiLU; writes f32 (for scan) and f16
// ---------------------------------------------------------------------------
__global__ void __launch_bounds__(256)
conv_silu_k(const float* __restrict__ xz, const float* __restrict__ cw,
            const float* __restrict__ cb, float* __restrict__ uc,
            _Float16* __restrict__ uc16)
{
    const int idx = blockIdx.x * 256 + threadIdx.x;
    if (idx >= NB * SEQ * DI) return;
    const int c  = idx & (DI - 1);
    const int l  = (idx >> 10) & (SEQ - 1);
    const int bb = idx >> 20;
    const size_t t = (size_t)bb * SEQ + l;

    const float w0 = cw[c * 3 + 0], w1 = cw[c * 3 + 1], w2 = cw[c * 3 + 2];
    const float um = (l > 0)       ? xz[(t - 1) * (2 * DI) + c] : 0.0f;
    const float u0 = xz[t * (2 * DI) + c];
    const float up = (l < SEQ - 1) ? xz[(t + 1) * (2 * DI) + c] : 0.0f;
    const float v  = w0 * um + w1 * u0 + w2 * up + cb[c];
    const float y  = v * sigmoidf_(v);
    uc[t * DI + c]   = y;
    uc16[t * DI + c] = (_Float16)y;
}

// ---------------------------------------------------------------------------
// Selective-scan: thread per (batch, d_inner); ym = (y + u*Dp)*silu(z) as f16
// ---------------------------------------------------------------------------
__global__ void __launch_bounds__(256)
ssm_scan_k(const float* __restrict__ delta, const float* __restrict__ bc,
           const float* __restrict__ uc, const float* __restrict__ xz,
           const float* __restrict__ alog, const float* __restrict__ Dp,
           _Float16* __restrict__ ym16)
{
    const int idx = blockIdx.x * 256 + threadIdx.x;
    if (idx >= NB * DI) return;
    const int d  = idx & (DI - 1);
    const int bb = idx >> 10;

    float A[16], h[16];
#pragma unroll
    for (int n = 0; n < 16; ++n) { A[n] = -__expf(alog[d * 16 + n]); h[n] = 0.0f; }
    const float Dv = Dp[d];

    for (int l = 0; l < SEQ; ++l) {
        const size_t t = (size_t)bb * SEQ + l;
        const float dt = delta[t * DI + d];
        const float u  = uc[t * DI + d];
        const float du = dt * u;
        float y = 0.0f;
#pragma unroll
        for (int n = 0; n < 16; ++n) {
            const float Bn = bc[t * 32 + n];
            const float Cn = bc[t * 32 + 16 + n];
            h[n] = __expf(dt * A[n]) * h[n] + du * Bn;
            y += h[n] * Cn;
        }
        const float z = xz[t * (2 * DI) + DI + d];
        ym16[t * DI + d] = (_Float16)((y + u * Dv) * (z * sigmoidf_(z)));
    }
}

// ---------------------------------------------------------------------------
// Flash attention (non-causal), f16 in / f16 out. Wave per (b,h,16-q tile).
// qkv16 layout (T,1536) = [q|k|v]; head h at dims 64h..64h+63.
// ---------------------------------------------------------------------------
__global__ void __launch_bounds__(32)
flash_attn_k(const _Float16* __restrict__ qkv, _Float16* __restrict__ out16)
{
    __shared__ _Float16 pls[16][32];

    const int lane = threadIdx.x;
    const int ml   = lane & 15;
    const int hi   = lane >> 4;
    const int qt   = blockIdx.x;
    const int bh   = blockIdx.y;
    const int b    = bh >> 3;
    const int h    = bh & 7;

    const _Float16* qp = qkv + ((size_t)(b * SEQ + qt * 16 + ml)) * 1536 + h * 64;
    v16h qa0 = make16(*(const v8h*)(qp + hi * 8), *(const v8h*)(qp + 16 + hi * 8));
    v16h qa1 = make16(*(const v8h*)(qp + 32 + hi * 8), *(const v8h*)(qp + 48 + hi * 8));
#pragma unroll
    for (int i = 0; i < 16; ++i) {           // fold 1/sqrt(dh) = 0.125 into Q
        qa0[i] *= (_Float16)0.125f;
        qa1[i] *= (_Float16)0.125f;
    }

    v8f o[4];
#pragma unroll
    for (int j = 0; j < 4; ++j) o[j] = v8f{};
    float mrun[8], drun[8];
#pragma unroll
    for (int v = 0; v < 8; ++v) { mrun[v] = -3.0e38f; drun[v] = 0.0f; }

    for (int kt = 0; kt < SEQ / 32; ++kt) {
        v16h kb[2][2];   // [d-chunk][key-subtile]
#pragma unroll
        for (int s = 0; s < 2; ++s)
#pragma unroll
            for (int c = 0; c < 2; ++c) {
                const _Float16* kp = qkv +
                    ((size_t)(b * SEQ + kt * 32 + s * 16 + ml)) * 1536 +
                    512 + h * 64 + c * 32 + hi * 8;
                kb[c][s] = make16(*(const v8h*)kp, *(const v8h*)(kp + 16));
            }

        v8f s0 = v8f{}, s1 = v8f{};
        s0 = wmma_f32_f16(qa0, kb[0][0], s0);
        s0 = wmma_f32_f16(qa1, kb[1][0], s0);
        s1 = wmma_f32_f16(qa0, kb[0][1], s1);
        s1 = wmma_f32_f16(qa1, kb[1][1], s1);

        float p0[8], p1[8], alpha[8];
#pragma unroll
        for (int v = 0; v < 8; ++v) {
            float mx = fmaxf(s0[v], s1[v]);
            mx = fmaxf(mx, __shfl_xor(mx, 1, 32));
            mx = fmaxf(mx, __shfl_xor(mx, 2, 32));
            mx = fmaxf(mx, __shfl_xor(mx, 4, 32));
            mx = fmaxf(mx, __shfl_xor(mx, 8, 32));
            const float nm = fmaxf(mrun[v], mx);
            alpha[v] = __expf(mrun[v] - nm);
            mrun[v]  = nm;
            p0[v] = __expf(s0[v] - nm);
            p1[v] = __expf(s1[v] - nm);
            float rs = p0[v] + p1[v];
            rs += __shfl_xor(rs, 1, 32);
            rs += __shfl_xor(rs, 2, 32);
            rs += __shfl_xor(rs, 4, 32);
            rs += __shfl_xor(rs, 8, 32);
            drun[v] = drun[v] * alpha[v] + rs;
        }
#pragma unroll
        for (int j = 0; j < 4; ++j)
#pragma unroll
            for (int v = 0; v < 8; ++v) o[j][v] *= alpha[v];

        __syncthreads();
#pragma unroll
        for (int v = 0; v < 8; ++v) {
            pls[hi * 8 + v][ml]      = (_Float16)p0[v];
            pls[hi * 8 + v][16 + ml] = (_Float16)p1[v];
        }
        __syncthreads();
        const _Float16* pr = &pls[ml][0];
        const v16h pa = make16(*(const v8h*)(pr + hi * 16), *(const v8h*)(pr + hi * 16 + 8));

#pragma unroll
        for (int j = 0; j < 4; ++j) {
            const _Float16* vp = qkv +
                ((size_t)(b * SEQ + kt * 32 + hi * 16)) * 1536 +
                1024 + h * 64 + j * 16 + ml;
            v16h vb;
#pragma unroll
            for (int i = 0; i < 16; ++i) vb[i] = vp[(size_t)i * 1536];
            o[j] = wmma_f32_f16(pa, vb, o[j]);
        }
        __syncthreads();
    }

#pragma unroll
    for (int v = 0; v < 8; ++v) {
        const float rcp = 1.0f / drun[v];
        const int row = b * SEQ + qt * 16 + hi * 8 + v;
#pragma unroll
        for (int j = 0; j < 4; ++j)
            out16[(size_t)row * DM + h * 64 + j * 16 + ml] = (_Float16)(o[j][v] * rcp);
    }
}

// ---------------------------------------------------------------------------
// Host launcher
// ---------------------------------------------------------------------------
static inline void cvt16(const float* in, _Float16* out, size_t n, hipStream_t s)
{
    cvt16_k<<<dim3((unsigned)((n + 255) / 256)), 256, 0, s>>>(in, out, (int)n);
}
static inline void launch_gemm(const _Float16* A16, const _Float16* W16,
                               const float* bias, const float* res,
                               float* C, _Float16* C16,
                               int M, int K, int N, int act, hipStream_t stream)
{
    dim3 grid((N + 63) / 64, M / 128);
    gemm_wmma_k<<<grid, 128, 0, stream>>>(A16, W16, bias, res, C, C16, K, N, act);
}

extern "C" void kernel_launch(void* const* d_in, const int* in_sizes, int n_in,
                              void* d_out, int out_size, void* d_ws, size_t ws_size,
                              hipStream_t stream)
{
    if (ws_size < WS_BYTES) return;

    char* ws = (char*)d_ws;
    float*     X     = (float*)(ws + B_X);
    float*     XZ    = (float*)(ws + B_XZ);
    float*     UC    = (float*)(ws + B_UC);
    float*     DL    = (float*)(ws + B_DL);
    float*     BC    = (float*)(ws + B_BC);
    _Float16*  LN16  = (_Float16*)(ws + B_LN16);
    _Float16*  UC16  = (_Float16*)(ws + B_UC16);
    _Float16*  YM16  = (_Float16*)(ws + B_YM16);
    _Float16*  X16   = (_Float16*)(ws + B_X16);
    _Float16*  TMP16 = (_Float16*)(ws + B_TMP16);
    _Float16*  ATT16 = (_Float16*)(ws + B_ATT16);
    _Float16*  W16   = (_Float16*)(ws + B_W16);

    (void)hipMemcpyAsync(X, d_in[0], (size_t)T_TOK * DM * sizeof(float),
                         hipMemcpyDeviceToDevice, stream);

    for (int layer = 0; layer < 6; ++layer) {
        const float* const* p = (const float* const*)(d_in + 1 + layer * 23);
        const float* in_proj   = p[0];
        const float* conv_w    = p[1];
        const float* conv_b    = p[2];
        const float* x_proj    = p[3];
        const float* dt_w      = p[4];
        const float* dt_b      = p[5];
        const float* out_proj  = p[6];
        const float* A_log     = p[7];
        const float* Dp        = p[8];
        const float* attn_in_w = p[9];
        const float* attn_in_b = p[10];
        const float* attn_ow   = p[11];
        const float* attn_ob   = p[12];
        const float* ffn_w1    = p[13];
        const float* ffn_b1    = p[14];
        const float* ffn_w2    = p[15];
        const float* ffn_b2    = p[16];
        const float* ln1w = p[17]; const float* ln1b = p[18];
        const float* ln2w = p[19]; const float* ln2b = p[20];
        const float* ln3w = p[21]; const float* ln3b = p[22];

        // convert this layer's GEMM weights to f16 (L2-resident, trivial cost)
        cvt16(in_proj,   W16 + W_INPROJ,  (size_t)2*DI*DM, stream);
        cvt16(dt_w,      W16 + W_DT,      (size_t)DI*DI,   stream);
        cvt16(x_proj,    W16 + W_XPROJ,   (size_t)32*DI,   stream);
        cvt16(out_proj,  W16 + W_OUTPROJ, (size_t)DM*DI,   stream);
        cvt16(attn_in_w, W16 + W_ATTNIN,  (size_t)3*DM*DM, stream);
        cvt16(attn_ow,   W16 + W_ATTNOUT, (size_t)DM*DM,   stream);
        cvt16(ffn_w1,    W16 + W_FFN1,    (size_t)FF*DM,   stream);
        cvt16(ffn_w2,    W16 + W_FFN2,    (size_t)DM*FF,   stream);

        // ---- Mamba block ----
        layernorm_k<<<T_TOK, 32, 0, stream>>>(X, nullptr, LN16, ln1w, ln1b);
        launch_gemm(LN16, W16 + W_INPROJ, nullptr, nullptr, XZ, nullptr,
                    T_TOK, DM, 2 * DI, 0, stream);
        conv_silu_k<<<(NB * SEQ * DI + 255) / 256, 256, 0, stream>>>(XZ, conv_w, conv_b, UC, UC16);
        launch_gemm(UC16, W16 + W_DT, dt_b, nullptr, DL, nullptr,
                    T_TOK, DI, DI, 2, stream);                                   // softplus
        launch_gemm(UC16, W16 + W_XPROJ, nullptr, nullptr, BC, nullptr,
                    T_TOK, DI, 32, 0, stream);
        ssm_scan_k<<<(NB * DI + 255) / 256, 256, 0, stream>>>(DL, BC, UC, XZ, A_log, Dp, YM16);
        launch_gemm(YM16, W16 + W_OUTPROJ, nullptr, X, X, X16,
                    T_TOK, DI, DM, 0, stream);                                   // X += mamba

        // ---- Attention block ----
        launch_gemm(X16, W16 + W_ATTNIN, attn_in_b, nullptr, nullptr, TMP16,
                    T_TOK, DM, 3 * DM, 0, stream);
        flash_attn_k<<<dim3(SEQ / 16, NB * NH), 32, 0, stream>>>(TMP16, ATT16);
        launch_gemm(ATT16, W16 + W_ATTNOUT, attn_ob, X, X, nullptr,
                    T_TOK, DM, DM, 0, stream);                                   // X += attn

        // ---- FFN block ----
        layernorm_k<<<T_TOK, 32, 0, stream>>>(X, X, nullptr, ln2w, ln2b);        // x = ln2(x)
        layernorm_k<<<T_TOK, 32, 0, stream>>>(X, nullptr, LN16, ln3w, ln3b);     // h = ln3(x)
        launch_gemm(LN16, W16 + W_FFN1, ffn_b1, nullptr, nullptr, TMP16,
                    T_TOK, DM, FF, 1, stream);                                   // gelu
        launch_gemm(TMP16, W16 + W_FFN2, ffn_b2, X, X, nullptr,
                    T_TOK, FF, DM, 0, stream);                                   // X += ffn
    }

    const float* lnw = (const float*)d_in[1 + 6 * 23];
    const float* lnb = (const float*)d_in[1 + 6 * 23 + 1];
    layernorm_k<<<T_TOK, 32, 0, stream>>>(X, (float*)d_out, nullptr, lnw, lnb);
}
